// DualPathSNNController_17609365914543
// MI455X (gfx1250) — matrix operand order repaired and checked by reading, not compiled, not run on hardware
//
#include <hip/hip_runtime.h>
#include <hip/hip_bf16.h>

typedef __attribute__((ext_vector_type(16))) _Float16 v16h;
typedef __attribute__((ext_vector_type(8)))  _Float16 v8h;
typedef __attribute__((ext_vector_type(8)))  float    v8f;

#define T_STEPS 1024
#define BATCH   2048
#define IN_DIM  9
#define H2      48
#define H_DIM   96
#define OUT_DIM 3
#define BETA_FAST 0.85f
#define BETA_SLOW 0.95f
#define BETA2     0.90f

// LDS row strides (in halves). 104h = 208B and 56h = 112B are 16B-aligned and
// give conflict-free bank patterns for the per-row ds_load_b128 A-fragment reads.
#define COMB_STRIDE 104
#define SPK2_STRIDE 56
#define N_WFRAG 11   // 9 x W2 fragments + 2 x Wout fragments staged in LDS

static __device__ __forceinline__ v8f wmma_f16f32(v16h a, v16h b, v8f c) {
  // D = A(16x32 f16) x B(32x16 f16) + C(16x16 f32)
  return __builtin_amdgcn_wmma_f32_16x16x32_f16(false, a, false, b, (short)0, c,
                                                false, false);
}

// Build a B-fragment (32x16 f16, K x N) from row-major W[nMaxRows][kDim].
// Lane layout: N = lane%16 ; lanes 0-15 hold K=kChunkBase+0..7 & +16..23, lanes 16-31
// hold K=kChunkBase+8..15 & +24..31 (halves 0..7 then 8..15).
static __device__ __forceinline__ v16h make_b_frag(const float* __restrict__ W,
                                                   int kDim, int kMax,
                                                   int nBase, int nMax,
                                                   int kChunkBase, int lane) {
  const int nn = nBase + (lane & 15);
  const int k0 = kChunkBase + ((lane >> 4) << 3);
  v16h b;
#pragma unroll
  for (int j = 0; j < 16; ++j) {
    const int k = k0 + j + ((j >> 3) << 3);
    float w = 0.0f;
    if (nn < nMax && k < kMax) w = W[nn * kDim + k];
    b[j] = (_Float16)w;
  }
  return b;
}

// Re-load a staged weight B-fragment from LDS (2 x ds_load_b128 per lane).
static __device__ __forceinline__ v16h load_w_frag(const _Float16* __restrict__ wbase,
                                                   int fidx, int lane) {
  const _Float16* p = wbase + ((fidx << 5) + lane) * 16;   // 32B per lane entry
  const v8h lo = *(const v8h*)p;
  const v8h hh = *(const v8h*)(p + 8);
  v16h b;
#pragma unroll
  for (int i = 0; i < 8; ++i) { b[i] = lo[i]; b[8 + i] = hh[i]; }
  return b;
}

// LIF carry folded into the WMMA C operand:
//   reset = (m > 1) ; C = beta*m + bias - reset ; m_new = A*B + C
static __device__ __forceinline__ v8f lif_carry(v8f m, float beta, float bias) {
  v8f c;
#pragma unroll
  for (int i = 0; i < 8; ++i) {
    const float reset = (m[i] > 1.0f) ? 1.0f : 0.0f;
    c[i] = beta * m[i] + bias - reset;
  }
  return c;
}

// Scatter a D-fragment's spikes (m_new > 1) into row-major LDS [row][col] f16.
// D layout: element i of lane l -> (M = i + 8*(l/16), N = l%16).
static __device__ __forceinline__ void store_spikes(_Float16* __restrict__ buf,
                                                    int stride, int colBase,
                                                    v8f m, int lane) {
  const int cc = colBase + (lane & 15);
  const int rbase = (lane >> 4) << 3;
#pragma unroll
  for (int i = 0; i < 8; ++i)
    buf[(rbase + i) * stride + cc] = (m[i] > 1.0f) ? (_Float16)1.0f : (_Float16)0.0f;
}

// A-fragment (16x32 f16) from row-major LDS. Lane l<16: row=l, K=kBase+0..7 &
// kBase+16..23 ; lane>=16: row=l-16, K=kBase+8..15 & kBase+24..31.
// Both 16B chunks are 16B-aligned -> ds_load_b128.
static __device__ __forceinline__ v16h load_a_frag(const _Float16* __restrict__ buf,
                                                   int stride, int kBase, int lane) {
  const int r = lane & 15;
  const int koff = kBase + ((lane >> 4) << 3);
  const _Float16* p = buf + r * stride + koff;
  const v8h lo = *(const v8h*)p;
  const v8h hh = *(const v8h*)(p + 16);
  v16h a;
#pragma unroll
  for (int i = 0; i < 8; ++i) { a[i] = lo[i]; a[8 + i] = hh[i]; }
  return a;
}

// Same, but the upper 16-K half is zero (used for the K=48->64 padded chunk).
static __device__ __forceinline__ v16h load_a_frag_lo(const _Float16* __restrict__ buf,
                                                      int stride, int kBase, int lane) {
  const int r = lane & 15;
  const int koff = kBase + ((lane >> 4) << 3);
  const v8h lo = *(const v8h*)(buf + r * stride + koff);
  v16h a;
#pragma unroll
  for (int i = 0; i < 8; ++i) { a[i] = lo[i]; a[8 + i] = (_Float16)0.0f; }
  return a;
}

// A-fragment from the binary input x[t] (K = 9 zero-padded to 32).
static __device__ __forceinline__ v16h load_a_x(const float* __restrict__ x,
                                                int t, int b0, int lane) {
  const int r = lane & 15;
  const float* xr = x + ((size_t)t * BATCH + (size_t)(b0 + r)) * IN_DIM;
  v16h a;
#pragma unroll
  for (int i = 0; i < 16; ++i) a[i] = (_Float16)0.0f;
  if (lane < 16) {           // K = 0..7 live in halves 0..7
#pragma unroll
    for (int j = 0; j < 8; ++j) a[j] = (_Float16)xr[j];
  } else {                   // K = 8 lives in half 0 of the upper lane group
    a[0] = (_Float16)xr[8];
  }
  return a;
}

__global__ __launch_bounds__(128, 1)
void snn_dualpath_kernel(const float* __restrict__ x,
                         const float* __restrict__ W1f, const float* __restrict__ b1f,
                         const float* __restrict__ W1s, const float* __restrict__ b1s,
                         const float* __restrict__ W2,  const float* __restrict__ b2,
                         const float* __restrict__ Wout,const float* __restrict__ bout,
                         float* __restrict__ out_rec,  float* __restrict__ mem_rec) {
  // Per-wave private staging buffers (no cross-wave sync needed in the loop).
  __shared__ __align__(16) _Float16 lds_comb[4][16 * COMB_STRIDE];
  __shared__ __align__(16) _Float16 lds_spk2[4][16 * SPK2_STRIDE];
  // Block-shared staged weight B-fragments: [frag][lane][16 halves] (32B/lane).
  __shared__ __align__(32) _Float16 lds_w[N_WFRAG][32][16];

  const int lane = threadIdx.x & 31;
  const int wave = threadIdx.x >> 5;
  const int tile = blockIdx.x * 4 + wave;   // 128 tiles of 16 batch rows
  const int b0 = tile * 16;
  const int n  = lane & 15;
  const int hi = lane >> 4;

  _Float16* comb = lds_comb[wave];
  _Float16* spk2 = lds_spk2[wave];

  // ---- Stage the once-per-step weights (W2, Wout) into LDS: one shared copy.
  if (wave == 0) {
#pragma unroll
    for (int ck = 0; ck < 3; ++ck)
#pragma unroll
      for (int j = 0; j < 3; ++j)
        *(v16h*)&lds_w[ck * 3 + j][lane][0] =
            make_b_frag(W2, H_DIM, H_DIM, j * 16, H2, ck * 32, lane);
    *(v16h*)&lds_w[9][lane][0]  = make_b_frag(Wout, H2, H2, 0, OUT_DIM, 0, lane);
    *(v16h*)&lds_w[10][lane][0] = make_b_frag(Wout, H2, H2, 0, OUT_DIM, 32, lane);
  }

  // ---- Layer-1 weights stay register-resident (top of each step's dep chain).
  v16h Bf[3], Bs[3];
#pragma unroll
  for (int j = 0; j < 3; ++j) {
    Bf[j] = make_b_frag(W1f, IN_DIM, IN_DIM, j * 16, H2, 0, lane);
    Bs[j] = make_b_frag(W1s, IN_DIM, IN_DIM, j * 16, H2, 0, lane);
  }

  // Per-lane biases (bias depends only on N = lane%16, splat across the C frag)
  float bf[3], bs[3], bb[3];
#pragma unroll
  for (int j = 0; j < 3; ++j) {
    bf[j] = b1f[j * 16 + n];
    bs[j] = b1s[j * 16 + n];
    bb[j] = b2[j * 16 + n];
  }
  const float bo = (n < OUT_DIM) ? bout[n] : 0.0f;

  // Membrane state (f32 fragments) stays in registers for the whole recurrence.
  v8f m1f[3], m1s[3], m2[3];
#pragma unroll
  for (int j = 0; j < 3; ++j)
#pragma unroll
    for (int i = 0; i < 8; ++i) { m1f[j][i] = 0.0f; m1s[j][i] = 0.0f; m2[j][i] = 0.0f; }

  __syncthreads();   // weight fragments visible to all 4 waves

#pragma unroll 1
  for (int t = 0; t < T_STEPS; ++t) {
    // Launder the LDS weight base so the loop-invariant ds_loads below cannot
    // be hoisted into registers (that would re-create >256 VGPR pressure).
    const _Float16* wbase = &lds_w[0][0][0];
    asm volatile("" : "+v"(wbase));

    // Prefetch next timestep's input rows (global_prefetch_b8).
    {
      const int tn = (t + 1 < T_STEPS) ? (t + 1) : t;
      __builtin_prefetch(x + ((size_t)tn * BATCH + (size_t)(b0 + n)) * IN_DIM, 0, 1);
    }

    const v16h ax = load_a_x(x, t, b0, lane);

    // ---- Layer 1 (fast + slow paths): 6 WMMAs, LIF folded into C.
#pragma unroll
    for (int j = 0; j < 3; ++j) {
      m1f[j] = wmma_f16f32(ax, Bf[j], lif_carry(m1f[j], BETA_FAST, bf[j]));
      store_spikes(comb, COMB_STRIDE, j * 16, m1f[j], lane);
      m1s[j] = wmma_f16f32(ax, Bs[j], lif_carry(m1s[j], BETA_SLOW, bs[j]));
      store_spikes(comb, COMB_STRIDE, 48 + j * 16, m1s[j], lane);
    }

    // ---- Layer 2: combined spikes [16x96] back from LDS as A-frags (in-order
    // same-wave LDS RAW), 9 chained WMMAs with W2 frags streamed from LDS.
    v16h a2[3];
#pragma unroll
    for (int ck = 0; ck < 3; ++ck) a2[ck] = load_a_frag(comb, COMB_STRIDE, ck * 32, lane);

#pragma unroll
    for (int j = 0; j < 3; ++j) {
      v8f acc = lif_carry(m2[j], BETA2, bb[j]);
      acc   = wmma_f16f32(a2[0], load_w_frag(wbase, 0 * 3 + j, lane), acc);
      acc   = wmma_f16f32(a2[1], load_w_frag(wbase, 1 * 3 + j, lane), acc);
      m2[j] = wmma_f16f32(a2[2], load_w_frag(wbase, 2 * 3 + j, lane), acc);
      store_spikes(spk2, SPK2_STRIDE, j * 16, m2[j], lane);

      // mem_rec[t, b0+row, j*16+n] — 16 consecutive lanes write 64B contiguous.
      const int rbase = hi << 3;
#pragma unroll
      for (int i = 0; i < 8; ++i)
        mem_rec[((size_t)t * BATCH + (size_t)(b0 + rbase + i)) * H2 + j * 16 + n] = m2[j][i];
    }

    // ---- Output layer: spk2[16x48] @ Wout^T (K padded 48->64): 2 WMMAs.
    const v16h ao0 = load_a_frag(spk2, SPK2_STRIDE, 0, lane);
    const v16h ao1 = load_a_frag_lo(spk2, SPK2_STRIDE, 32, lane);
    v8f co;
#pragma unroll
    for (int i = 0; i < 8; ++i) co[i] = bo;
    v8f dvec = wmma_f16f32(ao0, load_w_frag(wbase, 9, lane), co);
    dvec     = wmma_f16f32(ao1, load_w_frag(wbase, 10, lane), dvec);

    if (n < OUT_DIM) {
      const int rbase = hi << 3;
#pragma unroll
      for (int i = 0; i < 8; ++i)
        out_rec[((size_t)t * BATCH + (size_t)(b0 + rbase + i)) * OUT_DIM + n] = dvec[i];
    }
  }
}

extern "C" void kernel_launch(void* const* d_in, const int* in_sizes, int n_in,
                              void* d_out, int out_size, void* d_ws, size_t ws_size,
                              hipStream_t stream) {
  const float* x    = (const float*)d_in[0];
  const float* W1f  = (const float*)d_in[1];
  const float* b1f  = (const float*)d_in[2];
  const float* W1s  = (const float*)d_in[3];
  const float* b1s  = (const float*)d_in[4];
  const float* W2   = (const float*)d_in[5];
  const float* b2   = (const float*)d_in[6];
  const float* Wout = (const float*)d_in[7];
  const float* bout = (const float*)d_in[8];

  float* out_rec = (float*)d_out;                                   // [T,B,3]
  float* mem_rec = out_rec + (size_t)T_STEPS * BATCH * OUT_DIM;     // [T,B,48]

  dim3 grid(BATCH / (16 * 4));   // 32 blocks x 4 waves = 128 tiles of 16 rows
  dim3 block(128);
  hipLaunchKernelGGL(snn_dualpath_kernel, grid, block, 0, stream,
                     x, W1f, b1f, W1s, b1s, W2, b2, Wout, bout, out_rec, mem_rec);
}
